// LSTM_51204600103202
// MI455X (gfx1250) — matrix operand order repaired and checked by reading, not compiled, not run on hardware
//
#include <hip/hip_runtime.h>

typedef _Float16 v16h __attribute__((ext_vector_type(16)));
typedef _Float16 h2   __attribute__((ext_vector_type(2)));
typedef float    v8f  __attribute__((ext_vector_type(8)));

#define HIDDEN 128
#define SEQ 7
#define BATCH 65536
#define HS 132            // padded LDS row stride in halves (avoids bank conflicts)
#define TILES_PER_WG 8

#if __has_builtin(__builtin_amdgcn_tanhf)
// gfx1250 has native v_tanh_f32: 1 transcendental per activation.
__device__ __forceinline__ float tanh_fast(float x) {
  return __builtin_amdgcn_tanhf(x);
}
__device__ __forceinline__ float sigmoid_fast(float x) {
  return __builtin_fmaf(0.5f, __builtin_amdgcn_tanhf(0.5f * x), 0.5f);
}
#else
__device__ __forceinline__ float fast_exp(float x) {
  return __builtin_amdgcn_exp2f(x * 1.4426950408889634f);
}
__device__ __forceinline__ float sigmoid_fast(float x) {
  return __builtin_amdgcn_rcpf(1.0f + fast_exp(-x));
}
__device__ __forceinline__ float tanh_fast(float x) {
  return 2.0f * __builtin_amdgcn_rcpf(1.0f + fast_exp(-2.0f * x)) - 1.0f;
}
#endif

__global__ __launch_bounds__(256, 1)
void lstm_wmma_kernel(const float* __restrict__ x,
                      const float* __restrict__ x0,
                      const float* __restrict__ W_ih,
                      const float* __restrict__ W_hh,
                      const float* __restrict__ b_ih,
                      const float* __restrict__ b_hh,
                      const float* __restrict__ W_out,
                      const float* __restrict__ b_out,
                      float* __restrict__ out)
{
  __shared__ __align__(16) _Float16 h_lds[16 * HS];  // h state, f16, [M=16][n=128(+pad)]
  __shared__ float s_xs[SEQ][16];                    // x inputs for this batch tile
  __shared__ float s_pred[16];                       // per-row prediction accumulator

  const int tid  = threadIdx.x;
  const int wave = tid >> 5;
  const int lane = tid & 31;
  const int ln   = lane & 15;
  const int hi   = lane >> 4;        // which 16-lane half
  const int n0   = wave << 4;        // this wave's hidden-column tile
  const int col  = n0 + ln;          // hidden column 0..127

  // ---- per-lane constants: output weight, fused bias, input weight per gate ----
  const float wout = W_out[col];
  float biasg[4], wih[4];
#pragma unroll
  for (int g = 0; g < 4; ++g) {
    const int ng = g * HIDDEN + col;       // gate row in [4H]
    biasg[g] = b_ih[ng] + b_hh[ng];
    wih[g]   = W_ih[ng];
  }

  // ---- load B fragments (W_hh^T tiles) into registers once ----
  // Fragment (kk,g): K = kk*32..+31, N = gate g, cols n0..n0+15.
  // Lane map: n = col, k-half = hi; slot e (0..15) -> K = kk*32 + hi*8 + (e<8 ? e : e+8).
  v16h bfrag[16];
#pragma unroll
  for (int kk = 0; kk < 4; ++kk) {
#pragma unroll
    for (int g = 0; g < 4; ++g) {
      const int ng = g * HIDDEN + col;                       // W_hh row (gate output)
      const float* src = W_hh + ng * HIDDEN + kk * 32 + hi * 8;
      v16h bf;
#pragma unroll
      for (int e = 0; e < 16; ++e) bf[e] = (_Float16)src[e < 8 ? e : e + 8];
      bfrag[kk * 4 + g] = bf;
    }
  }

  const int tile0 = blockIdx.x * TILES_PER_WG;
  for (int bt = 0; bt < TILES_PER_WG; ++bt) {
    const int batch_base = (tile0 + bt) << 4;

    // zero h state (h0 = 0), stage x for this tile, zero pred accumulator
    {
      unsigned int* hz = (unsigned int*)h_lds;
#pragma unroll 2
      for (int i = tid; i < (16 * HS) / 2; i += 256) hz[i] = 0u;
      if (tid < SEQ * 16) {
        const int tt = tid >> 4, b = tid & 15;
        s_xs[tt][b] = (tt == 0) ? x0[batch_base + b]
                                : x[(batch_base + b) * SEQ + (tt - 1)];
      }
      if (tid < 16) s_pred[tid] = 0.0f;
    }

    v8f c_st;                                    // cell state, C-matrix layout
#pragma unroll
    for (int r = 0; r < 8; ++r) c_st[r] = 0.0f;

    for (int t = 0; t < SEQ; ++t) {
      __syncthreads();   // h_lds ready from previous step; s_pred zeroed

      // ---- load A fragments (h, 16x128 f16) from LDS ----
      // Lane map: M = ln, k-half = hi; slot e -> K = kk*32 + hi*8 + (e<8 ? e : e+8)
      v16h a[4];
#pragma unroll
      for (int kk = 0; kk < 4; ++kk) {
#pragma unroll
        for (int v = 0; v < 8; ++v) {
          const int e = 2 * v;
          const int K = kk * 32 + hi * 8 + (e < 8 ? e : e + 8);
          h2 p = *(const h2*)(&h_lds[ln * HS + K]);
          a[kk][e]     = p.x;
          a[kk][e + 1] = p.y;
        }
      }

      // ---- seed accumulators: bias + x_t * W_ih (input dim is 1) ----
      v8f acc[4];
#pragma unroll
      for (int g = 0; g < 4; ++g) {
#pragma unroll
        for (int r = 0; r < 8; ++r)
          acc[g][r] = __builtin_fmaf(s_xs[t][r + 8 * hi], wih[g], biasg[g]);
      }

      // ---- gates += h @ W_hh^T : 16 WMMAs, 4 independent chains ----
#pragma unroll
      for (int kk = 0; kk < 4; ++kk) {
#pragma unroll
        for (int g = 0; g < 4; ++g) {
          acc[g] = __builtin_amdgcn_wmma_f32_16x16x32_f16(
              false, a[kk], false, bfrag[kk * 4 + g], (short)0, acc[g],
              false, false);
        }
      }

      // ---- elementwise cell update (all in registers, C layout) ----
      float hv[8], pp[8];
#pragma unroll
      for (int r = 0; r < 8; ++r) {
        const float ig = sigmoid_fast(acc[0][r]);
        const float fg = sigmoid_fast(acc[1][r]);
        const float gg = tanh_fast(acc[2][r]);
        const float og = sigmoid_fast(acc[3][r]);
        const float cn = __builtin_fmaf(fg, c_st[r], ig * gg);
        c_st[r] = cn;
        hv[r] = og * tanh_fast(cn);
        pp[r] = hv[r] * wout;                 // pred partial for row r+8*hi
      }

      // ---- reduce pred partials across the 16 lanes of each half ----
#pragma unroll
      for (int r = 0; r < 8; ++r) {
        pp[r] += __shfl_xor(pp[r], 1, 32);
        pp[r] += __shfl_xor(pp[r], 2, 32);
        pp[r] += __shfl_xor(pp[r], 4, 32);
        pp[r] += __shfl_xor(pp[r], 8, 32);
      }
      if (ln == 0) {
#pragma unroll
        for (int r = 0; r < 8; ++r)
          atomicAdd(&s_pred[r + 8 * hi], pp[r]);
      }

      __syncthreads();   // A-fragment reads + pred atomics complete

      // ---- publish new h (f16) for next step's A operand ----
#pragma unroll
      for (int r = 0; r < 8; ++r)
        h_lds[(r + 8 * hi) * HS + col] = (_Float16)hv[r];

      // ---- one thread per batch row: emit prediction, re-zero accumulator ----
      if (tid < 16) {
        out[(batch_base + tid) * SEQ + t] = s_pred[tid] + b_out[0];
        s_pred[tid] = 0.0f;
      }
    }
    __syncthreads();     // drain before next tile reinitializes LDS
  }
}

extern "C" void kernel_launch(void* const* d_in, const int* in_sizes, int n_in,
                              void* d_out, int out_size, void* d_ws, size_t ws_size,
                              hipStream_t stream) {
  (void)in_sizes; (void)n_in; (void)d_ws; (void)ws_size; (void)out_size;
  const float* x     = (const float*)d_in[0];
  const float* x0    = (const float*)d_in[1];
  const float* W_ih  = (const float*)d_in[2];
  const float* W_hh  = (const float*)d_in[3];
  const float* b_ih  = (const float*)d_in[4];
  const float* b_hh  = (const float*)d_in[5];
  const float* W_out = (const float*)d_in[6];
  const float* b_out = (const float*)d_in[7];
  float* out = (float*)d_out;

  const int grid = BATCH / (16 * TILES_PER_WG);   // 512 workgroups
  lstm_wmma_kernel<<<grid, 256, 0, stream>>>(x, x0, W_ih, W_hh, b_ih, b_hh,
                                             W_out, b_out, out);
}
